// GNN_47794396070377
// MI455X (gfx1250) — matrix operand order repaired and checked by reading, not compiled, not run on hardware
//
#include <hip/hip_runtime.h>

#define N_NODES 100000
#define N_EDGES 1600000

typedef __attribute__((ext_vector_type(16))) _Float16 v16h;
typedef __attribute__((ext_vector_type(8)))  _Float16 v8h;
typedef __attribute__((ext_vector_type(8)))  float    v8f;

// ---------------- degree / inverse-count ----------------
__global__ void degree_kernel(const int* __restrict__ dst, float* __restrict__ cnt, int E) {
  int t = blockIdx.x * blockDim.x + threadIdx.x;
  if (t >= E) return;
  int d = dst[t];
  d = (d < 0) ? 0 : (d >= N_NODES ? N_NODES - 1 : d);
  atomicAdd(&cnt[d], 1.0f);
}

__global__ void inv_kernel(const float* __restrict__ cnt, float* __restrict__ inv, int n) {
  int t = blockIdx.x * blockDim.x + threadIdx.x;
  if (t >= n) return;
  inv[t] = 1.0f / fmaxf(cnt[t], 1.0f);
}

// ---------------- edge scatter: agg[dst] += X[src] ----------------
template <int LOG2C>
__global__ void scatter_kernel(const float* __restrict__ X, const int* __restrict__ src,
                               const int* __restrict__ dst, float* __restrict__ agg) {
  constexpr int C = 1 << LOG2C;     // chunks per edge
  constexpr int D = C * 4;          // feature dim
  long tid = (long)blockIdx.x * blockDim.x + threadIdx.x;
  const long total = (long)N_EDGES << LOG2C;
  if (tid >= total) return;
  long e = tid >> LOG2C;
  int  c = (int)(tid & (C - 1));
  int s = src[e]; s = (s < 0) ? 0 : (s >= N_NODES ? N_NODES - 1 : s);
  int t = dst[e]; t = (t < 0) ? 0 : (t >= N_NODES ? N_NODES - 1 : t);
  const float4 v = *(const float4*)(X + (long)s * D + c * 4);
  float* o = agg + (long)t * D + c * 4;
  atomicAdd(o + 0, v.x);
  atomicAdd(o + 1, v.y);
  atomicAdd(o + 2, v.z);
  atomicAdd(o + 3, v.w);
}

// ---------------- pack A_cat = [mean | x] in f16, row-major [N, 2D] ----------------
template <int D>
__global__ void packA_kernel(const float* __restrict__ agg, const float* __restrict__ inv,
                             const float* __restrict__ X, _Float16* __restrict__ Acat) {
  long tid = (long)blockIdx.x * blockDim.x + threadIdx.x;
  if (tid >= (long)N_NODES * D) return;
  long i = tid / D;
  int  j = (int)(tid - i * D);
  float m = agg[tid] * inv[i];
  _Float16* row = Acat + i * (2 * D);
  row[j]     = (_Float16)m;
  row[D + j] = (_Float16)X[tid];
}

// ---------------- pack W_cat = [Wl; Wr] into WMMA B-fragment order ----------------
// Packed half index = ((ntile*nKb + kb)*32 + lane)*16; lane holds column
// N = ntile*16 + (lane&15), K = kb*32 + (lane<16 ? 0 : 16) + i. ntile-major, so a
// 4-tile column group is one contiguous chunk (easy block-level LDS staging).
__global__ void packW_kernel(const float* __restrict__ Wl, const float* __restrict__ Wr,
                             _Float16* __restrict__ Wp, int d, int dout) {
  int Kt = 2 * d;
  int total = Kt * dout;
  int tid = blockIdx.x * blockDim.x + threadIdx.x;
  if (tid >= total) return;
  int i    = tid & 15;
  int lane = (tid >> 4) & 31;
  int rest = tid >> 9;
  int nKb  = Kt >> 5;
  int kb    = rest % nKb;
  int ntile = rest / nKb;
  int K = (kb << 5) + ((lane & 16) ? 16 : 0) + i;
  int n = (ntile << 4) + (lane & 15);
  float v = (K < d) ? Wl[(long)K * dout + n] : Wr[(long)(K - d) * dout + n];
  Wp[tid] = (_Float16)v;
}

// ---------------- WMMA GEMM: Y = relu(A_cat @ W_packed + bias) ----------------
// Block = 8 waves over 8 row tiles, all sharing one 64-column group whose B
// fragments are staged in LDS once per block. K-loop: 2 global loads (A) +
// 8 ds_load_b128 (B) + 4 independent WMMA chains -> LOADcnt tracks only the A
// stream (deep pipelining), B hits LDS latency via DScnt.
template <int KT, int DOUT>
__global__ __launch_bounds__(256) void gemm_kernel(const _Float16* __restrict__ A,
                                                   const _Float16* __restrict__ Wp,
                                                   const float* __restrict__ bias,
                                                   float* __restrict__ Y) {
  constexpr int NKB = KT / 32;        // K blocks of 32
  constexpr int NRT = N_NODES / 16;   // 6250 row tiles
  constexpr int BHALVES = 4 * NKB * 32 * 16;   // halves for 4 column tiles
  __shared__ __align__(16) _Float16 Bs[BHALVES];

  const int colGrp = blockIdx.y;       // group of 4 column tiles (64 cols)

  // ---- cooperative stage of the 4-tile B slice (contiguous in Wp) ----
  {
    const _Float16* wsrc = Wp + (long)colGrp * BHALVES;
    for (int idx = threadIdx.x; idx < BHALVES / 8; idx += 256)
      *(v8h*)(Bs + idx * 8) = *(const v8h*)(wsrc + idx * 8);
  }
  __syncthreads();

  const int lane = threadIdx.x & 31;
  const int wave = threadIdx.x >> 5;
  const int rowTile = blockIdx.x * 8 + wave;
  if (rowTile >= NRT) return;          // wave-uniform: EXEC stays all-ones
  const int r  = lane & 15;
  const int hi = lane >> 4;

  const _Float16* aptr = A + (long)(rowTile * 16 + r) * KT + hi * 16;
  const _Float16* bbase = Bs + (long)lane * 16;
  constexpr int BSTRIDE = NKB * 32 * 16;   // halves between column tiles in LDS

  v8f acc0 = {}, acc1 = {}, acc2 = {}, acc3 = {};
#pragma unroll
  for (int kb = 0; kb < NKB; ++kb) {
    // A fragment: lane<16 -> K = kb*32 + {0..15}; lane>=16 -> +16
    v8h alo = *(const v8h*)(aptr + kb * 32);
    v8h ahi = *(const v8h*)(aptr + kb * 32 + 8);
    v16h a;
#pragma unroll
    for (int q = 0; q < 8; ++q) { a[q] = alo[q]; a[q + 8] = ahi[q]; }
    const _Float16* bk = bbase + kb * 32 * 16;
    v16h b0 = *(const v16h*)(bk);
    v16h b1 = *(const v16h*)(bk + BSTRIDE);
    v16h b2 = *(const v16h*)(bk + 2 * BSTRIDE);
    v16h b3 = *(const v16h*)(bk + 3 * BSTRIDE);
    acc0 = __builtin_amdgcn_wmma_f32_16x16x32_f16(false, a, false, b0, (short)0, acc0,
                                                  false, false);
    acc1 = __builtin_amdgcn_wmma_f32_16x16x32_f16(false, a, false, b1, (short)0, acc1,
                                                  false, false);
    acc2 = __builtin_amdgcn_wmma_f32_16x16x32_f16(false, a, false, b2, (short)0, acc2,
                                                  false, false);
    acc3 = __builtin_amdgcn_wmma_f32_16x16x32_f16(false, a, false, b3, (short)0, acc3,
                                                  false, false);
  }

  const int mbase = rowTile * 16 + hi * 8;
  float* yrow = Y + (long)mbase * DOUT + colGrp * 64 + r;
  v8f accs[4] = {acc0, acc1, acc2, acc3};
#pragma unroll
  for (int j = 0; j < 4; ++j) {
    const float bv = bias[colGrp * 64 + j * 16 + r];
#pragma unroll
    for (int v = 0; v < 8; ++v) {     // C/D layout: VGPR v -> row M = v (+8 for hi lanes)
      float o = accs[j][v] + bv;
      o = o > 0.0f ? o : 0.0f;
      yrow[(long)v * DOUT + j * 16] = o;
    }
  }
}

// ---------------- driver ----------------
extern "C" void kernel_launch(void* const* d_in, const int* in_sizes, int n_in,
                              void* d_out, int out_size, void* d_ws, size_t ws_size,
                              hipStream_t stream) {
  (void)in_sizes; (void)n_in; (void)out_size; (void)ws_size;
  const float* x   = (const float*)d_in[0];
  const int*   ei  = (const int*)d_in[1];
  const int*   src = ei;
  const int*   dst = ei + N_EDGES;
  const float* Wl1 = (const float*)d_in[2];
  const float* Wr1 = (const float*)d_in[3];
  const float* b1  = (const float*)d_in[4];
  const float* Wl2 = (const float*)d_in[5];
  const float* Wr2 = (const float*)d_in[6];
  const float* b2  = (const float*)d_in[7];
  const float* Wl3 = (const float*)d_in[8];
  const float* Wr3 = (const float*)d_in[9];
  const float* b3  = (const float*)d_in[10];

  char* base = (char*)d_ws;
  size_t off = 0;
  auto take = [&](size_t bytes) {
    char* q = base + off;
    off += (bytes + 255) & ~(size_t)255;
    return (void*)q;
  };
  float*    agg  = (float*)take((size_t)N_NODES * 256 * 4);
  _Float16* Acat = (_Float16*)take((size_t)N_NODES * 512 * 2);
  float*    h    = (float*)take((size_t)N_NODES * 256 * 4);
  float*    cnt  = (float*)take((size_t)N_NODES * 4);
  float*    inv  = (float*)take((size_t)N_NODES * 4);
  _Float16* Wp1  = (_Float16*)take((size_t)256 * 256 * 2);
  _Float16* Wp2  = (_Float16*)take((size_t)512 * 256 * 2);
  _Float16* Wp3  = (_Float16*)take((size_t)512 * 128 * 2);

  const int B = 256;

  // degrees (shared by all layers)
  hipMemsetAsync(cnt, 0, (size_t)N_NODES * 4, stream);
  degree_kernel<<<(N_EDGES + B - 1) / B, B, 0, stream>>>(dst, cnt, N_EDGES);
  inv_kernel<<<(N_NODES + B - 1) / B, B, 0, stream>>>(cnt, inv, N_NODES);

  // weight packing (tiny)
  packW_kernel<<<(2 * 128 * 256 + B - 1) / B, B, 0, stream>>>(Wl1, Wr1, Wp1, 128, 256);
  packW_kernel<<<(2 * 256 * 256 + B - 1) / B, B, 0, stream>>>(Wl2, Wr2, Wp2, 256, 256);
  packW_kernel<<<(2 * 256 * 128 + B - 1) / B, B, 0, stream>>>(Wl3, Wr3, Wp3, 256, 128);

  // ---- Layer 1: d=128 -> 256 ----
  hipMemsetAsync(agg, 0, (size_t)N_NODES * 128 * 4, stream);
  {
    long total = (long)N_EDGES * 32;
    scatter_kernel<5><<<(unsigned)((total + B - 1) / B), B, 0, stream>>>(x, src, dst, agg);
  }
  {
    long total = (long)N_NODES * 128;
    packA_kernel<128><<<(unsigned)((total + B - 1) / B), B, 0, stream>>>(agg, inv, x, Acat);
  }
  {
    dim3 g((N_NODES / 16 + 7) / 8, 256 / 64);
    gemm_kernel<256, 256><<<g, 256, 0, stream>>>(Acat, Wp1, b1, h);
  }

  // ---- Layer 2: 256 -> 256 ----
  hipMemsetAsync(agg, 0, (size_t)N_NODES * 256 * 4, stream);
  {
    long total = (long)N_EDGES * 64;
    scatter_kernel<6><<<(unsigned)((total + B - 1) / B), B, 0, stream>>>(h, src, dst, agg);
  }
  {
    long total = (long)N_NODES * 256;
    packA_kernel<256><<<(unsigned)((total + B - 1) / B), B, 0, stream>>>(agg, inv, h, Acat);
  }
  {
    dim3 g((N_NODES / 16 + 7) / 8, 256 / 64);
    gemm_kernel<512, 256><<<g, 256, 0, stream>>>(Acat, Wp2, b2, h);
  }

  // ---- Layer 3: 256 -> 128 (output) ----
  hipMemsetAsync(agg, 0, (size_t)N_NODES * 256 * 4, stream);
  {
    long total = (long)N_EDGES * 64;
    scatter_kernel<6><<<(unsigned)((total + B - 1) / B), B, 0, stream>>>(h, src, dst, agg);
  }
  {
    long total = (long)N_NODES * 256;
    packA_kernel<256><<<(unsigned)((total + B - 1) / B), B, 0, stream>>>(agg, inv, h, Acat);
  }
  {
    dim3 g((N_NODES / 16 + 7) / 8, 128 / 64);
    gemm_kernel<512, 128><<<g, 256, 0, stream>>>(Acat, Wp3, b3, (float*)d_out);
  }
}
